// Model_82566451298423
// MI455X (gfx1250) — matrix operand order repaired and checked by reading, not compiled, not run on hardware
//
#include <hip/hip_runtime.h>

typedef __bf16 bf16_t;
typedef bf16_t v16bf __attribute__((ext_vector_type(16)));
typedef bf16_t v8bf  __attribute__((ext_vector_type(8)));
typedef bf16_t v2bf  __attribute__((ext_vector_type(2)));
typedef float  v8f   __attribute__((ext_vector_type(8)));
typedef float  v4f   __attribute__((ext_vector_type(4)));

#define NB    16
#define SQ    2048
#define DD    128
#define QB    128            // queries per workgroup (8 waves x 16)
#define KT    32             // k-tile
#define NW    8              // waves per workgroup
#define YSTR  136            // y_lds row stride (bf16 elems)
#define XSTR  40             // xT_lds row stride
#define PSTR  40             // p_lds row stride

#define LOG2E   1.4426950408889634f
#define QSCALE  (0.08838834764831845f * 1.4426950408889634f)

#if __has_builtin(__builtin_amdgcn_exp2f)
#define EXP2F(x) __builtin_amdgcn_exp2f(x)
#else
#define EXP2F(x) exp2f(x)
#endif

static_assert(sizeof(v16bf) == 32, "v16bf must be 8 VGPRs");
static_assert(sizeof(v8f)   == 32, "v8f must be 8 VGPRs");

__device__ __forceinline__ v8bf cvt8v(v4f a, v4f b) {
  v8bf r;
  r[0] = (bf16_t)a.x; r[1] = (bf16_t)a.y; r[2] = (bf16_t)a.z; r[3] = (bf16_t)a.w;
  r[4] = (bf16_t)b.x; r[5] = (bf16_t)b.y; r[6] = (bf16_t)b.z; r[7] = (bf16_t)b.w;
  return r;
}

__device__ __forceinline__ v8bf cvt8s(const float* s, float scale) {
  v4f a = *(const v4f*)s;
  v4f b = *(const v4f*)(s + 4);
  v8bf r;
  r[0] = (bf16_t)(a.x * scale); r[1] = (bf16_t)(a.y * scale);
  r[2] = (bf16_t)(a.z * scale); r[3] = (bf16_t)(a.w * scale);
  r[4] = (bf16_t)(b.x * scale); r[5] = (bf16_t)(b.y * scale);
  r[6] = (bf16_t)(b.z * scale); r[7] = (bf16_t)(b.w * scale);
  return r;
}

__device__ __forceinline__ v16bf pack16(v8bf lo, v8bf hi) {
  v16bf r;
#pragma unroll
  for (int j = 0; j < 8; ++j) { r[j] = lo[j]; r[8 + j] = hi[j]; }
  return r;
}

// cvt staged registers -> bf16 tiles in LDS (y row-major, x transposed as b32 pairs)
__device__ __forceinline__ void stage_commit(bf16_t* yb, bf16_t* xb,
    int yrow, int ycol, int kp, int db,
    v4f y0, v4f y1, v4f y2, v4f y3,
    v4f xa, v4f xbv, v4f xc, v4f xd)
{
  *(v8bf*)&yb[yrow * YSTR + ycol]     = cvt8v(y0, y1);
  *(v8bf*)&yb[yrow * YSTR + ycol + 8] = cvt8v(y2, y3);
#pragma unroll
  for (int j = 0; j < 8; ++j) {
    const float e0 = (j < 4) ? xa[j] : xbv[j - 4];
    const float e1 = (j < 4) ? xc[j] : xd[j - 4];
    v2bf p; p[0] = (bf16_t)e0; p[1] = (bf16_t)e1;
    *(v2bf*)&xb[(db + j) * XSTR + 2 * kp] = p;     // xT[d][2kp..2kp+1]
  }
}

__global__ void __launch_bounds__(256)
fattn_wmma(const float* __restrict__ X, const float* __restrict__ Y,
           const int* __restrict__ M32, float* __restrict__ Z)
{
  __shared__ bf16_t y_lds[2][KT * YSTR];      // double-buffered 32x128 K-tile
  __shared__ bf16_t xT_lds[2][DD * XSTR];     // double-buffered 128x32 V-tile (transposed)
  __shared__ bf16_t p_lds[NW * 16 * PSTR];    // per-wave 16x32 P scratch (wave-private)

  const int tid  = threadIdx.x;
  const int lane = tid & 31;
  const int wave = tid >> 5;
  const int b    = blockIdx.x >> 4;
  const int q0   = ((blockIdx.x & 15) * QB) + wave * 16;
  const int n16  = lane & 15;
  const int hi   = lane >> 4;

  // ---- Q fragments (A-layout), scale*log2e folded in
  v16bf aQ[4];
  {
    const float* qrow = X + ((size_t)b * SQ + (size_t)(q0 + n16)) * DD;
    const int lo = hi * 8;
#pragma unroll
    for (int c = 0; c < 4; ++c)
      aQ[c] = pack16(cvt8s(qrow + c * 32 + lo, QSCALE),
                     cvt8s(qrow + c * 32 + 16 + lo, QSCALE));
  }

  v8f acc[8];
#pragma unroll
  for (int f = 0; f < 8; ++f) acc[f] = (v8f){0.f, 0.f, 0.f, 0.f, 0.f, 0.f, 0.f, 0.f};
  float lrow[8];
#pragma unroll
  for (int r = 0; r < 8; ++r) lrow[r] = 0.0f;

  // staging maps
  const int yrow = tid >> 3;           // y stage: tile row 0..31
  const int ycol = (tid & 7) * 16;     // y stage: 16-float chunk
  const int kp   = tid & 15;           // x stage: k-pair -> k = 2*kp
  const int db   = (tid >> 4) * 8;     // x stage: 8-d group

  bf16_t* pbase = &p_lds[wave * 16 * PSTR];

  // ---- prologue: load + commit tile 0
  v4f sy0, sy1, sy2, sy3, sxa, sxb, sxc, sxd;
  {
    const float* ys = Y + ((size_t)b * SQ + (size_t)yrow) * DD + ycol;
    sy0 = *(const v4f*)ys;       sy1 = *(const v4f*)(ys + 4);
    sy2 = *(const v4f*)(ys + 8); sy3 = *(const v4f*)(ys + 12);
    const float* x0 = X + ((size_t)b * SQ + (size_t)(2 * kp)) * DD + db;
    sxa = *(const v4f*)x0;        sxb = *(const v4f*)(x0 + 4);
    sxc = *(const v4f*)(x0 + DD); sxd = *(const v4f*)(x0 + DD + 4);
  }
  stage_commit(&y_lds[0][0], &xT_lds[0][0], yrow, ycol, kp, db,
               sy0, sy1, sy2, sy3, sxa, sxb, sxc, sxd);
  __syncthreads();

  for (int kt = 0; kt < SQ; kt += KT) {
    const int  pb   = (kt >> 5) & 1;
    bf16_t*    yc   = &y_lds[pb][0];
    bf16_t*    xc   = &xT_lds[pb][0];
    const bool more = (kt + KT) < SQ;

    // ---- issue next tile's global loads early (consumed at bottom of iter)
    if (more) {
      const float* ys = Y + ((size_t)b * SQ + (size_t)(kt + KT + yrow)) * DD + ycol;
      sy0 = *(const v4f*)ys;       sy1 = *(const v4f*)(ys + 4);
      sy2 = *(const v4f*)(ys + 8); sy3 = *(const v4f*)(ys + 12);
      const float* x0 = X + ((size_t)b * SQ + (size_t)(kt + KT + 2 * kp)) * DD + db;
      sxa = *(const v4f*)x0;        sxb = *(const v4f*)(x0 + 4);
      sxc = *(const v4f*)(x0 + DD); sxd = *(const v4f*)(x0 + DD + 4);
    }

    // ---- mask -> C-layout seed (log2 domain)
    v8f sc[2];
#pragma unroll
    for (int h = 0; h < 2; ++h) {
#pragma unroll
      for (int r = 0; r < 8; ++r) {
        const int    qr = q0 + r + 8 * hi;
        const size_t mi = ((size_t)b * SQ + (size_t)qr) * SQ
                        + (size_t)(kt + 16 * h + n16);
        sc[h][r] = (float)__builtin_nontemporal_load(M32 + 2 * mi) * LOG2E;
      }
    }

    // ---- scores (log2 domain): S = (Q*qscale).Ytile^T + mask*log2e
#pragma unroll
    for (int h = 0; h < 2; ++h) {
      const bf16_t* br = &yc[(h * 16 + n16) * YSTR + hi * 8];
      v8f s = sc[h];
#pragma unroll
      for (int c = 0; c < 4; ++c) {
        v16bf bb = pack16(*(const v8bf*)(br + c * 32), *(const v8bf*)(br + c * 32 + 16));
        s = __builtin_amdgcn_wmma_f32_16x16x32_bf16(false, aQ[c], false, bb,
                                                    (short)0, s, false, false);
      }
      sc[h] = s;
    }

    // ---- e = 2^s; per-lane partial row sums; P -> wave-private LDS (no barrier needed)
#pragma unroll
    for (int r = 0; r < 8; ++r) {
      const float e0 = EXP2F(sc[0][r]);
      const float e1 = EXP2F(sc[1][r]);
      lrow[r] += e0 + e1;
      bf16_t* pr = pbase + (r + 8 * hi) * PSTR;
      pr[n16]      = (bf16_t)e0;
      pr[n16 + 16] = (bf16_t)e1;
    }

    // ---- z += P(16x32) . Xtile(32x128); same-wave DS ordering covers p_lds RAW
    v16bf aP;
    {
      const bf16_t* pr = pbase + n16 * PSTR + hi * 8;
      aP = pack16(*(const v8bf*)(pr), *(const v8bf*)(pr + 16));
    }
#pragma unroll
    for (int f = 0; f < 8; ++f) {
      const bf16_t* br = &xc[(f * 16 + n16) * XSTR + hi * 8];
      v16bf bb = pack16(*(const v8bf*)(br), *(const v8bf*)(br + 16));
      acc[f] = __builtin_amdgcn_wmma_f32_16x16x32_bf16(false, aP, false, bb,
                                                       (short)0, acc[f], false, false);
    }

    // ---- commit next tile into the other buffer; single barrier per iteration
    if (more)
      stage_commit(&y_lds[pb ^ 1][0], &xT_lds[pb ^ 1][0], yrow, ycol, kp, db,
                   sy0, sy1, sy2, sy3, sxa, sxb, sxc, sxd);
    __syncthreads();
  }

  // ---- end-of-kernel row-sum reduction (16-lane halves), then normalize + store
  float rinv[8];
#pragma unroll
  for (int r = 0; r < 8; ++r) {
    float rs = lrow[r];
#pragma unroll
    for (int d = 1; d < 16; d <<= 1) rs += __shfl_xor(rs, d, 32);
    rinv[r] = 1.0f / rs;
  }
#pragma unroll
  for (int f = 0; f < 8; ++f) {
#pragma unroll
    for (int r = 0; r < 8; ++r) {
      const int qr = q0 + r + 8 * hi;
      Z[((size_t)b * SQ + (size_t)qr) * DD + f * 16 + n16] = acc[f][r] * rinv[r];
    }
  }
}

extern "C" void kernel_launch(void* const* d_in, const int* in_sizes, int n_in,
                              void* d_out, int out_size, void* d_ws, size_t ws_size,
                              hipStream_t stream) {
  (void)in_sizes; (void)n_in; (void)out_size; (void)d_ws; (void)ws_size;
  const float* x = (const float*)d_in[0];
  const float* y = (const float*)d_in[1];
  const int*   m = (const int*)d_in[2];   // int64 mask viewed as dword pairs (LE low word)
  float*       z = (float*)d_out;
  dim3 grid(NB * (SQ / QB));              // 256 workgroups
  fattn_wmma<<<grid, 256, 0, stream>>>(x, y, m, z);
}
